// AsymmetricCGAAttention_30288109371533
// MI455X (gfx1250) — matrix operand order repaired and checked by reading, not compiled, not run on hardware
//
#include <hip/hip_runtime.h>
#include <hip/hip_bf16.h>

typedef _Float16 h16;
typedef __attribute__((ext_vector_type(16))) _Float16 v16h;
typedef __attribute__((ext_vector_type(8)))  _Float16 v8h;
typedef __attribute__((ext_vector_type(8)))  float    v8f;

#define NTOK 21
#define Dm   128
#define NH   2
#define DHd  64
#define D2   32
#define NP   10
#define NS   11
#define EPSv 1e-8f
#define SCALE 0.125f

// ---- workspace layout (bytes) ----
#define OFF_WQKV 0
#define OFF_WOUT (OFF_WQKV + 4*24*32*16*2)   // 98304
#define OFF_WG   (OFF_WOUT + 4*8*32*16*2)
#define OFF_WR   (OFF_WG   + 4*8*32*16*2)
#define OFF_PART (OFF_WR   + 4*8*32*16*2)
#define OFF_MEAN (OFF_PART + 512*4)

// ---- LDS layout (bytes) ----
#define L_X     0
#define L_QKV   10752
#define L_SC    59904
#define L_SC2   68096
#define L_Y2P   70144
#define L_GATE  78336
#define L_OUT   86528
#define L_NORM  97280
#define L_F16   97408
#define L_H16   97408
#define L_Q16   105600
#define L_K16   113792
#define L_VT    121984
#define L_VST   130176
#define L_P16   138368
#define L_P2    142464
#define L_Y16   144512
#define L_Y2    152704
#define LDS_BYTES 156800

__device__ __forceinline__ float wave_sum(float v) {
#pragma unroll
  for (int m = 16; m >= 1; m >>= 1) v += __shfl_xor(v, m, 32);
  return v;
}
__device__ __forceinline__ float wave_max(float v) {
#pragma unroll
  for (int m = 16; m >= 1; m >>= 1) v = fmaxf(v, __shfl_xor(v, m, 32));
  return v;
}
__device__ __forceinline__ v16h combine16(v8h lo, v8h hi) {
  v16h r;
#pragma unroll
  for (int i = 0; i < 8; ++i) { r[i] = lo[i]; r[i+8] = hi[i]; }
  return r;
}
// A-fragment (16-bit 16x32): lane row = tile_row + lane%16; 16 contiguous K halfs at 2*ab = (lane>=16)?16:0
__device__ __forceinline__ v16h ldA(const h16* rowp, int ab) {
  return combine16(*(const v8h*)(rowp + 2*ab), *(const v8h*)(rowp + 2*ab + 8));
}
// B-fragment (16-bit 32x16): lane col = lane%16; 16 contiguous K halfs at kb = (lane>=16)?16:0
__device__ __forceinline__ v16h ldB(const h16* p) {
  return combine16(*(const v8h*)p, *(const v8h*)(p + 8));
}
__device__ __forceinline__ v8f wmma16(v16h a, v16h b, v8f c) {
  return __builtin_amdgcn_wmma_f32_16x16x32_f16(false, a, false, b, (short)0, c, false, false);
}
__device__ __forceinline__ float sigm(float z) { return 1.f / (1.f + __expf(-z)); }

// ---------------- K0: weights -> f16 WMMA B-fragments ----------------
__global__ void k_prep_weights(const float* __restrict__ Wqkv, const float* __restrict__ Wout,
                               const float* __restrict__ Wg, const float* __restrict__ Wr,
                               h16* __restrict__ ws) {
  int bb = blockIdx.x;
  const float* src; h16* dst; int ncols, ntc, tile;
  if (bb < 96)       { src = Wqkv; dst = ws + OFF_WQKV/2; ncols = 384; tile = bb;      ntc = 24; }
  else if (bb < 128) { src = Wout; dst = ws + OFF_WOUT/2; ncols = 128; tile = bb - 96; ntc = 8; }
  else if (bb < 160) { src = Wg;   dst = ws + OFF_WG/2;   ncols = 128; tile = bb - 128; ntc = 8; }
  else               { src = Wr;   dst = ws + OFF_WR/2;   ncols = 128; tile = bb - 160; ntc = 8; }
  int kt = tile / ntc, nt = tile % ntc;
  for (int e = threadIdx.x; e < 512; e += 256) {
    int l = e >> 4, i = e & 15;
    int n  = nt * 16 + (l & 15);
    int kk = kt * 32 + ((l >= 16) ? 16 : 0) + i;
    dst[(tile * 32 + l) * 16 + i] = (h16)src[kk * ncols + n];
  }
}

// ---------------- K1: per-block partial sums of ||h_patch|| ----------------
__global__ void k_norm_partial(const float* __restrict__ x, const float* __restrict__ g,
                               float* __restrict__ partials) {
  __shared__ float red[8];
  int tid = threadIdx.x, lane = tid & 31, wave = tid >> 5;
  float g0 = g[lane], g1 = g[lane+32], g2 = g[lane+64], g3 = g[lane+96];
  float acc = 0.f;
  for (int i = 0; i < 32; ++i) {
    int r = blockIdx.x * 256 + wave * 32 + i;      // 0 .. 81919
    int b = r / NP, p = r % NP;
    const float* xr = x + ((long)b * NTOK + (2*p+1)) * Dm;
    float x0 = xr[lane], x1 = xr[lane+32], x2 = xr[lane+64], x3 = xr[lane+96];
    float mean = wave_sum(x0+x1+x2+x3) * (1.f/128.f);
    x0 -= mean; x1 -= mean; x2 -= mean; x3 -= mean;
    float ss = wave_sum(x0*x0 + x1*x1 + x2*x2 + x3*x3);
    float rinv = rsqrtf(ss * (1.f/128.f) + EPSv);
    float h0 = x0*rinv*g0, h1 = x1*rinv*g1, h2 = x2*rinv*g2, h3 = x3*rinv*g3;
    acc += sqrtf(wave_sum(h0*h0 + h1*h1 + h2*h2 + h3*h3));
  }
  if (lane == 0) red[wave] = acc;
  __syncthreads();
  if (wave == 0) {
    float v = (lane < 8) ? red[lane] : 0.f;
    v = wave_sum(v);
    if (lane == 0) partials[blockIdx.x] = v;
  }
}

// ---------------- K2: deterministic final reduce ----------------
__global__ void k_norm_reduce(const float* __restrict__ partials, float* __restrict__ mean_out) {
  __shared__ float red[256];
  int t = threadIdx.x;
  float v = 0.f;
  for (int i = t; i < 320; i += 256) v += partials[i];
  red[t] = v; __syncthreads();
  for (int s = 128; s > 0; s >>= 1) { if (t < s) red[t] += red[t + s]; __syncthreads(); }
  if (t == 0) mean_out[0] = red[0] / 81920.f;
}

// ---------------- main: one workgroup per batch element ----------------
__global__ void k_main(const float* __restrict__ x,
                       const float* __restrict__ fcr, const float* __restrict__ fci,
                       const float* __restrict__ g,  const float* __restrict__ bqkv,
                       const float* __restrict__ bout, const float* __restrict__ bg,
                       const float* __restrict__ br, const float* __restrict__ Bbias,
                       const h16* __restrict__ wq_fr, const h16* __restrict__ wo_fr,
                       const h16* __restrict__ wg_fr, const h16* __restrict__ wr_fr,
                       const float* __restrict__ mean_norm, float* __restrict__ out) {
  extern __shared__ __align__(16) unsigned char smem[];
  float* s_x    = (float*)(smem + L_X);
  float* s_qkv  = (float*)(smem + L_QKV);
  float* s_sc   = (float*)(smem + L_SC);
  float* s_sc2  = (float*)(smem + L_SC2);
  float* s_y2p  = (float*)(smem + L_Y2P);
  float* s_gate = (float*)(smem + L_GATE);
  float* s_out  = (float*)(smem + L_OUT);
  float* s_norm = (float*)(smem + L_NORM);
  h16* s_h16 = (h16*)(smem + L_H16);
  h16* s_q16 = (h16*)(smem + L_Q16);
  h16* s_k16 = (h16*)(smem + L_K16);
  h16* s_vT  = (h16*)(smem + L_VT);
  h16* s_vsT = (h16*)(smem + L_VST);
  h16* s_p16 = (h16*)(smem + L_P16);
  h16* s_p2  = (h16*)(smem + L_P2);
  h16* s_y16 = (h16*)(smem + L_Y16);
  h16* s_y2  = (h16*)(smem + L_Y2);

  const int tid = threadIdx.x, lane = tid & 31, wave = tid >> 5;
  const int b = blockIdx.x;
  const int lm = lane & 15;
  const int hiHalf = (lane >> 4) & 1;
  const int ab = hiHalf ? 8 : 0;     // A-frag K offset
  const int kb = hiHalf ? 16 : 0;    // B-frag K offset

  { // zero all f16 buffers (zero padding feeds zero through softmax / P@V)
    unsigned int* z = (unsigned int*)(smem + L_F16);
    for (int i = tid; i < (LDS_BYTES - L_F16) / 4; i += 256) z[i] = 0u;
  }
  __syncthreads();

  // phase 1: h = zcrmsnorm(x, g); keep x, h(f16), per-token ||h||
  {
    float g0 = g[lane], g1 = g[lane+32], g2 = g[lane+64], g3 = g[lane+96];
    for (int tok = wave; tok < NTOK; tok += 8) {
      const float* xr = x + ((long)b * NTOK + tok) * Dm;
      float x0 = xr[lane], x1 = xr[lane+32], x2 = xr[lane+64], x3 = xr[lane+96];
      s_x[tok*Dm+lane]=x0; s_x[tok*Dm+lane+32]=x1; s_x[tok*Dm+lane+64]=x2; s_x[tok*Dm+lane+96]=x3;
      float mean = wave_sum(x0+x1+x2+x3) * (1.f/128.f);
      x0 -= mean; x1 -= mean; x2 -= mean; x3 -= mean;
      float ss = wave_sum(x0*x0 + x1*x1 + x2*x2 + x3*x3);
      float rinv = rsqrtf(ss * (1.f/128.f) + EPSv);
      float h0 = x0*rinv*g0, h1 = x1*rinv*g1, h2 = x2*rinv*g2, h3 = x3*rinv*g3;
      float n2 = wave_sum(h0*h0 + h1*h1 + h2*h2 + h3*h3);
      if (lane == 0) s_norm[tok] = sqrtf(n2);
      s_h16[tok*Dm+lane]=(h16)h0; s_h16[tok*Dm+lane+32]=(h16)h1;
      s_h16[tok*Dm+lane+64]=(h16)h2; s_h16[tok*Dm+lane+96]=(h16)h3;
    }
  }
  __syncthreads();

  // phase 2: qkv = h @ Wqkv + bqkv   (2 x 24 tiles, K=128)
  for (int j = wave; j < 48; j += 8) {
    int mt = j / 24, nt = j % 24;
    v8f acc = {};
#pragma unroll
    for (int kt = 0; kt < 4; ++kt) {
      v16h a  = ldA(s_h16 + (mt*16 + lm)*Dm + kt*32, ab);
      v16h bf = ldB(wq_fr + ((kt*24 + nt)*32 + lane)*16);
      acc = wmma16(a, bf, acc);
    }
    int col = nt*16 + lm;
    float bias = bqkv[col];
#pragma unroll
    for (int r = 0; r < 8; ++r)
      s_qkv[(mt*16 + r + (hiHalf?8:0))*384 + col] = acc[r] + bias;
  }
  __syncthreads();

  // phase 3: RoPE q,k -> f16; V stored transposed (dim-major), plus compact stat-V
  for (int it = tid; it < NH*NTOK*D2; it += 256) {
    int h = it / (NTOK*D2), rem = it % (NTOK*D2);
    int tok = rem / D2, d = rem % D2;
    const float* qr = s_qkv + tok*384;
    float qa = qr[h*64+2*d],     qb = qr[h*64+2*d+1];
    float ka = qr[128+h*64+2*d], kbv = qr[128+h*64+2*d+1];
    float va = qr[256+h*64+2*d], vb = qr[256+h*64+2*d+1];
    float fr = fcr[tok*D2+d], fi = fci[tok*D2+d];
    s_q16[(h*32+tok)*64 + 2*d]   = (h16)(qa*fr - qb*fi);
    s_q16[(h*32+tok)*64 + 2*d+1] = (h16)(qa*fi + qb*fr);
    s_k16[(h*32+tok)*64 + 2*d]   = (h16)(ka*fr - kbv*fi);
    s_k16[(h*32+tok)*64 + 2*d+1] = (h16)(ka*fi + kbv*fr);
    s_vT[(h*64+2*d)*32 + tok]   = (h16)va;
    s_vT[(h*64+2*d+1)*32 + tok] = (h16)vb;
    if ((tok & 1) == 0) {
      int sk = tok >> 1;
      s_vsT[(h*64+2*d)*32 + sk]   = (h16)va;
      s_vsT[(h*64+2*d+1)*32 + sk] = (h16)vb;
    }
  }
  __syncthreads();

  // phase 4: scores (full: 8 tile-jobs; patch: waves 0/1, row/col gathers via per-lane addressing)
  {
    int h = wave >> 2, mt = (wave >> 1) & 1, nt = wave & 1;
    v8f acc = {};
#pragma unroll
    for (int kt = 0; kt < 2; ++kt) {
      v16h a  = ldA(s_q16 + (h*32 + mt*16 + lm)*64 + kt*32, ab);
      v16h bf = ldB(s_k16 + (h*32 + nt*16 + lm)*64 + kt*32 + kb);
      acc = wmma16(a, bf, acc);
    }
    int col = nt*16 + lm;
#pragma unroll
    for (int r = 0; r < 8; ++r) {
      int row = mt*16 + r + (hiHalf?8:0);
      float v = acc[r] * SCALE;
      if (row < NTOK && col < NTOK) v += Bbias[row*NTOK + col];
      s_sc[(h*32+row)*32 + col] = v;
    }
    if (wave < 2) {
      int hh = wave;
      int qrow = (lm < NP) ? (2*lm + 1) : 31;        // PATCH_POS gather (row 31 is zero pad)
      v8f acc2 = {};
#pragma unroll
      for (int kt = 0; kt < 2; ++kt) {
        v16h a  = ldA(s_q16 + (hh*32 + qrow)*64 + kt*32, ab);
        int krow = (lm < NS) ? (2*lm) : 31;          // STAT_POS gather
        v16h bf = ldB(s_k16 + (hh*32 + krow)*64 + kt*32 + kb);
        acc2 = wmma16(a, bf, acc2);
      }
#pragma unroll
      for (int r = 0; r < 8; ++r)
        s_sc2[(hh*16 + r + (hiHalf?8:0))*16 + lm] = acc2[r] * SCALE;
    }
  }
  __syncthreads();

  // phase 5: softmax (f16 probabilities; pad lanes write exact zeros)
  for (int r = wave; r < 42; r += 8) {
    int h = r / 21, m = r % 21;
    float v = (lane < NTOK) ? s_sc[(h*32+m)*32 + lane] : -1e30f;
    float mx = wave_max(v);
    float e = (lane < NTOK) ? __expf(v - mx) : 0.f;
    float s = wave_sum(e);
    s_p16[(h*32+m)*32 + lane] = (h16)(e / s);
  }
  for (int r = wave; r < 20; r += 8) {
    int h = r / 10, m = r % 10;
    float v = (lane < NS) ? s_sc2[(h*16+m)*16 + lane] : -1e30f;
    float mx = wave_max(v);
    float e = (lane < NS) ? __expf(v - mx) : 0.f;
    float s = wave_sum(e);
    s_p2[(h*16+m)*32 + lane] = (h16)(e / s);
  }
  __syncthreads();

  // phase 6: y_full = P @ V  (K=32, single WMMA per tile)
  for (int j = wave; j < 16; j += 8) {
    int h = j >> 3, mt = (j >> 2) & 1, nt = j & 3;
    v16h a  = ldA(s_p16 + (h*32 + mt*16 + lm)*32, ab);
    v16h bf = ldB(s_vT + (h*64 + nt*16 + lm)*32 + kb);
    v8f acc = {};
    acc = wmma16(a, bf, acc);
    int col = h*64 + nt*16 + lm;
#pragma unroll
    for (int r = 0; r < 8; ++r)
      s_y16[(mt*16 + r + (hiHalf?8:0))*Dm + col] = (h16)acc[r];
  }
  __syncthreads();

  // phase 7: s_out = x + y_full @ Wout + bout
  for (int j = wave; j < 16; j += 8) {
    int mt = j >> 3, nt = j & 7;
    v8f acc = {};
#pragma unroll
    for (int kt = 0; kt < 4; ++kt) {
      v16h a  = ldA(s_y16 + (mt*16 + lm)*Dm + kt*32, ab);
      v16h bf = ldB(wo_fr + ((kt*8 + nt)*32 + lane)*16);
      acc = wmma16(a, bf, acc);
    }
    int col = nt*16 + lm;
    float bias = bout[col];
#pragma unroll
    for (int r = 0; r < 8; ++r) {
      int row = mt*16 + r + (hiHalf?8:0);
      if (row < NTOK) s_out[row*Dm + col] = s_x[row*Dm + col] + acc[r] + bias;
    }
  }
  // phase 8a: y2 = P2 @ Vs (independent of phase 7 buffers; no barrier needed)
  {
    int h = wave >> 2, nt = wave & 3;
    v16h a  = ldA(s_p2 + (h*16 + lm)*32, ab);
    v16h bf = ldB(s_vsT + (h*64 + nt*16 + lm)*32 + kb);
    v8f acc = {};
    acc = wmma16(a, bf, acc);
    int col = h*64 + nt*16 + lm;
#pragma unroll
    for (int r = 0; r < 8; ++r)
      s_y2[(r + (hiHalf?8:0))*Dm + col] = (h16)acc[r];
  }
  // phase 8b: gate = sigmoid(h_patch @ Wg + bg) * (1 - sigmoid(||h|| - mean))
  {
    float mn = mean_norm[0];
    int nt = wave;
    int arow = (lm < NP) ? (2*lm + 1) : 31;          // PATCH_POS gather
    v8f acc = {};
#pragma unroll
    for (int kt = 0; kt < 4; ++kt) {
      v16h a  = ldA(s_h16 + arow*Dm + kt*32, ab);
      v16h bf = ldB(wg_fr + ((kt*8 + nt)*32 + lane)*16);
      acc = wmma16(a, bf, acc);
    }
    int col = nt*16 + lm;
    float bias = bg[col];
#pragma unroll
    for (int r = 0; r < 8; ++r) {
      int row = r + (hiHalf?8:0);
      int prow = (row < NP) ? (2*row + 1) : 1;
      float unc = 1.f - sigm(s_norm[prow] - mn);
      s_gate[row*Dm + col] = sigm(acc[r] + bias) * unc;
    }
  }
  __syncthreads();

  // phase 9: y2p = y2 @ Wr + br
  {
    int nt = wave;
    v8f acc = {};
#pragma unroll
    for (int kt = 0; kt < 4; ++kt) {
      v16h a  = ldA(s_y2 + lm*Dm + kt*32, ab);
      v16h bf = ldB(wr_fr + ((kt*8 + nt)*32 + lane)*16);
      acc = wmma16(a, bf, acc);
    }
    int col = nt*16 + lm;
    float bias = br[col];
#pragma unroll
    for (int r = 0; r < 8; ++r)
      s_y2p[(r + (hiHalf?8:0))*Dm + col] = acc[r] + bias;
  }
  __syncthreads();

  // phase 10: fused patch-add + writeout (odd rows < 21 are exactly PATCH_POS)
  for (int it = tid; it < NTOK*Dm; it += 256) {
    int tok = it / Dm, col = it % Dm;
    float v = s_out[it];
    if (tok & 1) {
      int p = tok >> 1;
      v += s_gate[p*Dm + col] * s_y2p[p*Dm + col];
    }
    out[((long)b * NTOK + tok) * Dm + col] = v;
  }
}

extern "C" void kernel_launch(void* const* d_in, const int* in_sizes, int n_in,
                              void* d_out, int out_size, void* d_ws, size_t ws_size,
                              hipStream_t stream) {
  (void)in_sizes; (void)n_in; (void)out_size; (void)ws_size;
  const float* x    = (const float*)d_in[0];
  const float* fcr  = (const float*)d_in[1];
  const float* fci  = (const float*)d_in[2];
  const float* g    = (const float*)d_in[3];
  const float* Wqkv = (const float*)d_in[4];
  const float* bqkv = (const float*)d_in[5];
  const float* Wout = (const float*)d_in[6];
  const float* bout = (const float*)d_in[7];
  const float* Wg   = (const float*)d_in[8];
  const float* bg   = (const float*)d_in[9];
  const float* Wr   = (const float*)d_in[10];
  const float* br   = (const float*)d_in[11];
  const float* Bb   = (const float*)d_in[12];
  float* out = (float*)d_out;
  unsigned char* ws = (unsigned char*)d_ws;
  float* partials = (float*)(ws + OFF_PART);
  float* meanp    = (float*)(ws + OFF_MEAN);

  k_prep_weights<<<192, 256, 0, stream>>>(Wqkv, Wout, Wg, Wr, (h16*)ws);
  k_norm_partial<<<320, 256, 0, stream>>>(x, g, partials);
  k_norm_reduce<<<1, 256, 0, stream>>>(partials, meanp);
  k_main<<<8192, 256, LDS_BYTES, stream>>>(x, fcr, fci, g, bqkv, bout, bg, br, Bb,
      (const h16*)(ws + OFF_WQKV), (const h16*)(ws + OFF_WOUT),
      (const h16*)(ws + OFF_WG),   (const h16*)(ws + OFF_WR),
      meanp, out);
}
